// GumbelTreeGRU_42185168781906
// MI455X (gfx1250) — matrix operand order repaired and checked by reading, not compiled, not run on hardware
//
#include <hip/hip_runtime.h>
#include <hip/hip_bf16.h>
#include <math.h>

// ---------------------------------------------------------------------------
// GumbelTreeGRU on MI455X (gfx1250): bf16 WMMA GEMMs + f32 state/gates.
// B=512, MAX_LEN=64, H=512, HQ=51.
// v4: scoring head moved onto WMMA (Wq1 zero-padded 51->64 cols, in-register
//     q.Wq2 contraction via 16-lane butterflies); NH bf16 mirror emitted in
//     GEMM2 epilogue; division-free h->bf16 mirror.
// ---------------------------------------------------------------------------

#define BATCH 512
#define MAXLEN 64
#define HDIM 512
#define HQDIM 51
#define HQPAD 64
#define K1 1024   // 2H
#define N1 1536   // 3H

typedef __attribute__((ext_vector_type(16))) __bf16 v16bf;
typedef __attribute__((ext_vector_type(8)))  __bf16 v8bf;
typedef __attribute__((ext_vector_type(8)))  float  v8f;

__device__ __forceinline__ float sigmoidf_(float x) {
  return 1.0f / (1.0f + expf(-x));
}

__device__ __forceinline__ v16bf join8(v8bf a, v8bf b) {
  return __builtin_shufflevector(a, b, 0,1,2,3,4,5,6,7,8,9,10,11,12,13,14,15);
}

// A-fragment: lane holds K {k0..k0+15} contiguous
__device__ __forceinline__ v16bf loadAfrag(const __bf16* p) {
  v8bf s0 = *(const v8bf*)(p);
  v8bf s1 = *(const v8bf*)(p + 8);
  return join8(s0, s1);
}

// ---------------------------------------------------------------------------
// Elementwise f32 -> bf16 (for weights)
// ---------------------------------------------------------------------------
__global__ __launch_bounds__(256) void f32_to_bf16_kernel(
    const float* __restrict__ src, __bf16* __restrict__ dst, int nElem) {
  int i = blockIdx.x * blockDim.x + threadIdx.x;
  if (i < nElem) dst[i] = (__bf16)src[i];
}

// Padded scoring-head params: Wq1pad (64x512 bf16, rows >=51 zero),
// bq1pad/wq2pad (64 f32, entries >=51 zero).
__global__ __launch_bounds__(256) void prep_q_kernel(
    const float* __restrict__ Wq1, const float* __restrict__ bq1,
    const float* __restrict__ Wq2, __bf16* __restrict__ Wq1pad,
    float* __restrict__ bq1pad, float* __restrict__ wq2pad) {
  const int i = blockIdx.x * blockDim.x + threadIdx.x;
  if (i < HQPAD * HDIM) {
    const int r = i >> 9;
    Wq1pad[i] = (r < HQDIM) ? (__bf16)Wq1[i] : (__bf16)0.0f;
  }
  if (i < HQPAD) {
    bq1pad[i] = (i < HQDIM) ? bq1[i] : 0.0f;
    wq2pad[i] = (i < HQDIM) ? Wq2[i] : 0.0f;
  }
}

// h (f32, row stride MAXLEN) -> bf16 mirror, rows [0, L) per batch.
// Grid: ((L*HDIM+255)/256, BATCH)
__global__ __launch_bounds__(256) void h_to_bf16_kernel(
    const float* __restrict__ src, __bf16* __restrict__ dst, int per) {
  const int rem = blockIdx.x * blockDim.x + threadIdx.x;
  if (rem >= per) return;
  const size_t idx = (size_t)blockIdx.y * MAXLEN * HDIM + rem;
  dst[idx] = (__bf16)src[idx];
}

// ---------------------------------------------------------------------------
// GEMM1 fused:  V = cat1 @ W1^T + b1, split i|f|r, gates, emit cat2/P/SI.
// Grid: (M/128, 8).  Block: 128 M-rows x 64-col window (in each region).
// 8 waves = 4 M-quarters x 2 col-halves. Wave: 2 M-tiles x 2 N-tiles x 3 reg
// = 12 accumulators (96 VGPRs).
// ---------------------------------------------------------------------------
__global__ __launch_bounds__(256) void gemm1_fused_kernel(
    const __bf16* __restrict__ hbf, const float* __restrict__ hcur,
    const __bf16* __restrict__ W1bf, const float* __restrict__ b1,
    __bf16* __restrict__ cat2, float* __restrict__ P, float* __restrict__ SI,
    int n) {
  const int wave  = threadIdx.x >> 5;
  const int lane  = threadIdx.x & 31;
  const int lhalf = lane >> 4;
  const int lmod  = lane & 15;
  const int wq    = wave >> 1;                   // M quarter (0..3)
  const int wc    = wave & 1;                    // col half (0/1)
  const int mblk  = blockIdx.x * 128 + wq * 32;
  const int cg    = blockIdx.y * 64 + wc * 32;   // col base in [0,512)

  // Per-M-tile A row pointers (row = left child; right child is +HDIM).
  const __bf16* rowL[2];
#pragma unroll
  for (int mt = 0; mt < 2; ++mt) {
    const int m = mblk + mt * 16 + lmod;
    const int bb = m / n;
    const int jj = m - bb * n;
    rowL[mt] = hbf + ((size_t)(bb * MAXLEN + jj)) * HDIM + lhalf * 16;
  }

  // Per-(region,ntile) W1 row pointers (with per-lane K half offset).
  const __bf16* bptr[6];
#pragma unroll
  for (int r = 0; r < 3; ++r)
#pragma unroll
    for (int nt = 0; nt < 2; ++nt)
      bptr[r * 2 + nt] =
          W1bf + (size_t)(r * HDIM + cg + nt * 16 + lmod) * K1 + lhalf * 16;

  v8f acc[3][2][2];
#pragma unroll
  for (int r = 0; r < 3; ++r)
#pragma unroll
    for (int mt = 0; mt < 2; ++mt)
#pragma unroll
      for (int nt = 0; nt < 2; ++nt) acc[r][mt][nt] = (v8f)0.0f;

  // K loop split into cat1 halves (left/right child rows of h).
#pragma unroll
  for (int half = 0; half < 2; ++half) {
    const __bf16* a0p = rowL[0] + half * HDIM;
    const __bf16* a1p = rowL[1] + half * HDIM;
    const int kw = half * HDIM;
#pragma unroll 2
    for (int kk = 0; kk < HDIM; kk += 32) {
      v16bf a0 = loadAfrag(a0p + kk);
      v16bf a1 = loadAfrag(a1p + kk);
      v16bf bq[6];
#pragma unroll
      for (int q = 0; q < 6; ++q) bq[q] = *(const v16bf*)(bptr[q] + kw + kk);
#pragma unroll
      for (int r = 0; r < 3; ++r)
#pragma unroll
        for (int nt = 0; nt < 2; ++nt) {
          acc[r][0][nt] = __builtin_amdgcn_wmma_f32_16x16x32_bf16(
              false, a0, false, bq[r * 2 + nt], (short)0, acc[r][0][nt],
              false, false);
          acc[r][1][nt] = __builtin_amdgcn_wmma_f32_16x16x32_bf16(
              false, a1, false, bq[r * 2 + nt], (short)0, acc[r][1][nt],
              false, false);
        }
    }
  }

  // Epilogue. C element (g,lane): M = tile + g + lhalf*8, col = tile + lmod.
  float bi[2], bfv[2], brv[2];
#pragma unroll
  for (int nt = 0; nt < 2; ++nt) {
    const int crel = cg + nt * 16 + lmod;
    bi[nt]  = b1[crel];
    bfv[nt] = b1[HDIM + crel];
    brv[nt] = b1[2 * HDIM + crel];
  }
#pragma unroll
  for (int mt = 0; mt < 2; ++mt) {
    const int mr0 = mblk + mt * 16 + lhalf * 8;  // row at g=0
    int b2 = mr0 / n;
    int j2 = mr0 - b2 * n;
#pragma unroll
    for (int g = 0; g < 8; ++g) {
      const size_t hrow = ((size_t)(b2 * MAXLEN + j2)) * HDIM;
      const size_t mrow = (size_t)(mr0 + g);
#pragma unroll
      for (int nt = 0; nt < 2; ++nt) {
        const int crel = cg + nt * 16 + lmod;
        const float iv = acc[0][mt][nt][g] + bi[nt];
        const float fv = acc[1][mt][nt][g] + bfv[nt];
        const float rv = acc[2][mt][nt][g] + brv[nt];
        const float rs = sigmoidf_(rv);
        const float hl = hcur[hrow + crel];
        const float hr = hcur[hrow + HDIM + crel];
        const size_t mo = mrow * HDIM + crel;
        cat2[mrow * K1 + crel]        = (__bf16)(hl * rs);
        cat2[mrow * K1 + HDIM + crel] = (__bf16)(hr * rs);
        P[mo]  = (hl + hr) * sigmoidf_(fv);
        SI[mo] = sigmoidf_(iv);
      }
      if (++j2 == n) { j2 = 0; ++b2; }  // advance (b,j) without division
    }
  }
}

// ---------------------------------------------------------------------------
// GEMM2 fused:  h_hat = cat2 @ W2^T + b2 ;  NH = P + tanh(h_hat) * SI
// Also writes NHbf (bf16 mirror) for the WMMA scoring head.
// Grid: (M/128, 8). Same wave arrangement; 4 accumulators per wave.
// ---------------------------------------------------------------------------
__global__ __launch_bounds__(256) void gemm2_fused_kernel(
    const __bf16* __restrict__ cat2, const __bf16* __restrict__ W2bf,
    const float* __restrict__ b2, const float* __restrict__ P,
    const float* __restrict__ SI, float* __restrict__ NH,
    __bf16* __restrict__ NHbf) {
  const int wave  = threadIdx.x >> 5;
  const int lane  = threadIdx.x & 31;
  const int lhalf = lane >> 4;
  const int lmod  = lane & 15;
  const int wq    = wave >> 1;
  const int wc    = wave & 1;
  const int mblk  = blockIdx.x * 128 + wq * 32;
  const int cg    = blockIdx.y * 64 + wc * 32;

  const __bf16* arow[2];
#pragma unroll
  for (int mt = 0; mt < 2; ++mt)
    arow[mt] = cat2 + (size_t)(mblk + mt * 16 + lmod) * K1 + lhalf * 16;

  const __bf16* bptr[2];
#pragma unroll
  for (int nt = 0; nt < 2; ++nt)
    bptr[nt] = W2bf + (size_t)(cg + nt * 16 + lmod) * K1 + lhalf * 16;

  v8f acc[2][2];
#pragma unroll
  for (int mt = 0; mt < 2; ++mt)
#pragma unroll
    for (int nt = 0; nt < 2; ++nt) acc[mt][nt] = (v8f)0.0f;

#pragma unroll 4
  for (int k = 0; k < K1; k += 32) {
    v16bf a0 = loadAfrag(arow[0] + k);
    v16bf a1 = loadAfrag(arow[1] + k);
    v16bf b0 = *(const v16bf*)(bptr[0] + k);
    v16bf b1v = *(const v16bf*)(bptr[1] + k);
    acc[0][0] = __builtin_amdgcn_wmma_f32_16x16x32_bf16(
        false, a0, false, b0, (short)0, acc[0][0], false, false);
    acc[0][1] = __builtin_amdgcn_wmma_f32_16x16x32_bf16(
        false, a0, false, b1v, (short)0, acc[0][1], false, false);
    acc[1][0] = __builtin_amdgcn_wmma_f32_16x16x32_bf16(
        false, a1, false, b0, (short)0, acc[1][0], false, false);
    acc[1][1] = __builtin_amdgcn_wmma_f32_16x16x32_bf16(
        false, a1, false, b1v, (short)0, acc[1][1], false, false);
  }

#pragma unroll
  for (int mt = 0; mt < 2; ++mt) {
#pragma unroll
    for (int nt = 0; nt < 2; ++nt) {
      const int c = cg + nt * 16 + lmod;
      const float bb = b2[c];
#pragma unroll
      for (int g = 0; g < 8; ++g) {
        const int mr = mblk + mt * 16 + g + lhalf * 8;
        const size_t mo = (size_t)mr * HDIM + c;
        const float hh = acc[mt][nt][g] + bb;
        const float nh = P[mo] + tanhf(hh) * SI[mo];
        NH[mo] = nh;
        NHbf[mo] = (__bf16)nh;
      }
    }
  }
}

// ---------------------------------------------------------------------------
// WMMA scoring head: q = tanh(NH @ Wq1pad^T + bq1pad);
// logit = q @ wq2pad + bq2  (padded cols 51..63 contribute zero).
// Grid: (M/128). Block: 8 waves; wave owns 16 rows x 64 (padded) cols.
// Final contraction: butterfly over each 16-lane half (C layout: lanes 0-15
// hold row g, lanes 16-31 hold row g+8).
// ---------------------------------------------------------------------------
__global__ __launch_bounds__(256) void logits_wmma_kernel(
    const __bf16* __restrict__ NHbf, const __bf16* __restrict__ Wq1pad,
    const float* __restrict__ bq1pad, const float* __restrict__ wq2pad,
    const float* __restrict__ bq2, float* __restrict__ logits) {
  const int wave  = threadIdx.x >> 5;
  const int lane  = threadIdx.x & 31;
  const int lhalf = lane >> 4;
  const int lmod  = lane & 15;
  const int mrow  = blockIdx.x * 128 + wave * 16;  // 16 rows per wave

  const __bf16* arow = NHbf + (size_t)(mrow + lmod) * HDIM + lhalf * 16;
  const __bf16* bp[4];
#pragma unroll
  for (int nt = 0; nt < 4; ++nt)
    bp[nt] = Wq1pad + (size_t)(nt * 16 + lmod) * HDIM + lhalf * 16;

  v8f acc[4];
#pragma unroll
  for (int nt = 0; nt < 4; ++nt) acc[nt] = (v8f)0.0f;

#pragma unroll 4
  for (int k = 0; k < HDIM; k += 32) {
    v16bf a = loadAfrag(arow + k);
#pragma unroll
    for (int nt = 0; nt < 4; ++nt) {
      v16bf bv = *(const v16bf*)(bp[nt] + k);
      acc[nt] = __builtin_amdgcn_wmma_f32_16x16x32_bf16(
          false, a, false, bv, (short)0, acc[nt], false, false);
    }
  }

  float bq1v[4], wq2v[4];
#pragma unroll
  for (int nt = 0; nt < 4; ++nt) {
    const int c = nt * 16 + lmod;
    bq1v[nt] = bq1pad[c];
    wq2v[nt] = wq2pad[c];
  }
#pragma unroll
  for (int g = 0; g < 8; ++g) {
    float partial = 0.0f;
#pragma unroll
    for (int nt = 0; nt < 4; ++nt)
      partial += tanhf(acc[nt][g] + bq1v[nt]) * wq2v[nt];
#pragma unroll
    for (int o = 1; o < 16; o <<= 1) partial += __shfl_xor(partial, o, 32);
    if (lmod == 0) logits[mrow + g + lhalf * 8] = partial + bq2[0];
  }
}

// ---------------------------------------------------------------------------
// Per-batch masked argmax (first-occurrence ties), select-mask output,
// row merge, and `done` (length) gating. Writes hnext rows [0, n).
// ---------------------------------------------------------------------------
__global__ __launch_bounds__(256) void select_combine_kernel(
    const float* __restrict__ hcur, const float* __restrict__ NH,
    const float* __restrict__ logits, const int* __restrict__ length,
    float* __restrict__ hnext, float* __restrict__ sel_out,
    int n, int step, int do_select) {
  const int b   = blockIdx.x;
  const int tid = threadIdx.x;
  __shared__ float sv[256];
  __shared__ int   sidx[256];
  __shared__ int   s_sel;

  const int len  = length[b];
  const int done = ((step + 1) < len) ? 1 : 0;
  int selidx = 0;

  if (do_select) {
    float bestv = -INFINITY;
    int   besti = 0x7fffffff;
    for (int j = tid; j < n; j += 256) {
      float lv = ((step + 1 + j) < len) ? logits[b * n + j] : -3.402823466e38f;
      if (lv > bestv) { bestv = lv; besti = j; }
    }
    sv[tid] = bestv; sidx[tid] = besti;
    __syncthreads();
    for (int s = 128; s > 0; s >>= 1) {
      if (tid < s) {
        float ov = sv[tid + s]; int oi = sidx[tid + s];
        if (ov > sv[tid] || (ov == sv[tid] && oi < sidx[tid])) {
          sv[tid] = ov; sidx[tid] = oi;
        }
      }
      __syncthreads();
    }
    if (tid == 0) s_sel = sidx[0];
    __syncthreads();
    selidx = s_sel;
    for (int j = tid; j < n; j += 256)
      sel_out[(size_t)b * n + j] = (j == selidx) ? 1.0f : 0.0f;
  }

  const float* hrow  = hcur  + (size_t)b * MAXLEN * HDIM;
  float*       orow  = hnext + (size_t)b * MAXLEN * HDIM;
  const float* nhrow = NH    + (size_t)b * n * HDIM;
  const int total = n * HDIM;
  for (int idx = tid; idx < total; idx += 256) {
    const int j = idx >> 9;
    const int c = idx & (HDIM - 1);
    const float hl = hrow[j * HDIM + c];
    float val;
    if (!done)            val = hl;
    else if (!do_select)  val = nhrow[j * HDIM + c];
    else if (j == selidx) val = nhrow[j * HDIM + c];
    else if (j <  selidx) val = hl;
    else                  val = hrow[(j + 1) * HDIM + c];
    orow[j * HDIM + c] = val;
  }
}

__global__ __launch_bounds__(256) void out_copy_kernel(
    const float* __restrict__ hfin, float* __restrict__ out) {
  const int i = blockIdx.x * blockDim.x + threadIdx.x;  // B*H
  const int b = i >> 9;
  const int c = i & (HDIM - 1);
  out[i] = hfin[(size_t)b * MAXLEN * HDIM + c];
}

// ---------------------------------------------------------------------------
// Host launch
// ---------------------------------------------------------------------------
extern "C" void kernel_launch(void* const* d_in, const int* in_sizes, int n_in,
                              void* d_out, int out_size, void* d_ws,
                              size_t ws_size, hipStream_t stream) {
  const float* input  = (const float*)d_in[0];
  const int*   length = (const int*)d_in[1];
  const float* W1     = (const float*)d_in[2];
  const float* b1     = (const float*)d_in[3];
  const float* W2     = (const float*)d_in[4];
  const float* b2     = (const float*)d_in[5];
  const float* Wq1    = (const float*)d_in[6];
  const float* bq1    = (const float*)d_in[7];
  const float* Wq2    = (const float*)d_in[8];
  const float* bq2    = (const float*)d_in[9];
  float* out = (float*)d_out;

  char* ws = (char*)d_ws;
  size_t off = 0;
  auto carve = [&](size_t bytes) -> void* {
    void* p = (void*)(ws + off);
    off = (off + bytes + 255) & ~(size_t)255;
    return p;
  };
  const size_t HTOT = (size_t)BATCH * MAXLEN * HDIM;
  const size_t MMAX = (size_t)BATCH * 63;
  __bf16* W1bf   = (__bf16*)carve((size_t)N1 * K1 * 2);
  __bf16* W2bf   = (__bf16*)carve((size_t)HDIM * K1 * 2);
  __bf16* Wq1pad = (__bf16*)carve((size_t)HQPAD * HDIM * 2);
  float*  bq1pad = (float*)carve(HQPAD * 4);
  float*  wq2pad = (float*)carve(HQPAD * 4);
  float*  hA     = (float*)carve(HTOT * 4);
  float*  hB     = (float*)carve(HTOT * 4);
  __bf16* hbf    = (__bf16*)carve(HTOT * 2);
  __bf16* cat2   = (__bf16*)carve(MMAX * K1 * 2);
  float*  Pb     = (float*)carve(MMAX * HDIM * 4);
  float*  SIb    = (float*)carve(MMAX * HDIM * 4);
  float*  NH     = (float*)carve(MMAX * HDIM * 4);
  __bf16* NHbf   = (__bf16*)carve(MMAX * HDIM * 2);
  float*  lg     = (float*)carve(MMAX * 4);
  (void)ws_size; (void)in_sizes; (void)n_in; (void)out_size;

  f32_to_bf16_kernel<<<(N1 * K1 + 255) / 256, 256, 0, stream>>>(W1, W1bf,
                                                                N1 * K1);
  f32_to_bf16_kernel<<<(HDIM * K1 + 255) / 256, 256, 0, stream>>>(W2, W2bf,
                                                                  HDIM * K1);
  prep_q_kernel<<<(HQPAD * HDIM + 255) / 256, 256, 0, stream>>>(
      Wq1, bq1, Wq2, Wq1pad, bq1pad, wq2pad);
  hipMemcpyAsync(hA, input, HTOT * 4, hipMemcpyDeviceToDevice, stream);

  float* hcur = hA;
  float* hnext = hB;
  size_t sel_off = (size_t)BATCH * HDIM;

  for (int i = 0; i < MAXLEN - 1; ++i) {
    const int n = (MAXLEN - 1) - i;      // pair count this step
    const int L = MAXLEN - i;            // live rows of h
    const int M = BATCH * n;             // rows; multiple of 512
    const int do_sel = (i < MAXLEN - 2) ? 1 : 0;

    h_to_bf16_kernel<<<dim3((L * HDIM + 255) / 256, BATCH), 256, 0, stream>>>(
        hcur, hbf, L * HDIM);
    gemm1_fused_kernel<<<dim3(M / 128, 8), 256, 0, stream>>>(
        hbf, hcur, W1bf, b1, cat2, Pb, SIb, n);
    gemm2_fused_kernel<<<dim3(M / 128, 8), 256, 0, stream>>>(
        cat2, W2bf, b2, Pb, SIb, NH, NHbf);
    if (do_sel) {
      logits_wmma_kernel<<<M / 128, 256, 0, stream>>>(NHbf, Wq1pad, bq1pad,
                                                      wq2pad, bq2, lg);
    }
    float* selp = do_sel ? (out + sel_off) : nullptr;
    select_combine_kernel<<<BATCH, 256, 0, stream>>>(hcur, NH, lg, length,
                                                     hnext, selp, n, i,
                                                     do_sel);
    if (do_sel) sel_off += (size_t)BATCH * n;
    float* tmp = hcur; hcur = hnext; hnext = tmp;
  }

  out_copy_kernel<<<(BATCH * HDIM) / 256, 256, 0, stream>>>(hcur, out);
}
